// WindowAttention_55113020342487
// MI455X (gfx1250) — matrix operand order repaired and checked by reading, not compiled, not run on hardware
//
#include <hip/hip_runtime.h>

typedef __attribute__((ext_vector_type(16))) __bf16 v16bf;
typedef __attribute__((ext_vector_type(8)))  float  v8f;

union FragU { uint4 u[2]; v16bf v; };

// pack two f32 -> two bf16 (round-half-up) in one dword via v_perm_b32
__device__ __forceinline__ unsigned int pack_bf16(float a, float b) {
  unsigned int ua = __float_as_uint(a) + 0x8000u;
  unsigned int ub = __float_as_uint(b) + 0x8000u;
  // D bytes (LSB..MSB) = {ua.b2, ua.b3, ub.b2, ub.b3}; S0=ub(codes 4-7), S1=ua(0-3)
  return __builtin_amdgcn_perm(ub, ua, 0x07060302u);
}

__device__ __forceinline__ unsigned short cvt_bf16(float a) {
  return (unsigned short)((__float_as_uint(a) + 0x8000u) >> 16);
}

// A-fragment (16x32 bf16) from row-major [M][K] storage, stride in halves.
// Per ISA 7.12.2: lane m=l&15; V0-3 hold K = (l<16?0:8)+0..7 at k0,
// V4-7 hold K = 16+(l<16?0:8)+0..7.
__device__ __forceinline__ v16bf load_frag_a(const unsigned short* base,
                                             int rowBase, int stride, int k0) {
  int lane = threadIdx.x & 31;
  int m = lane & 15;
  int kk = k0 + ((lane & 16) >> 1);           // +8 for upper half-wave
  const unsigned short* p = base + (size_t)(rowBase + m) * stride + kk;
  FragU f;
  f.u[0] = *(const uint4*)(p);                // halves k..k+7
  f.u[1] = *(const uint4*)(p + 16);           // halves k+16..k+23
  return f.v;
}

// B-fragment (32x16 bf16) from B^T stored row-major [N][K], stride in halves.
// Lane n=l&15; V0-3 hold K = (l<16?0:8)+0..7 at k0, V4-7 hold K = 16+(l<16?0:8)+0..7.
__device__ __forceinline__ v16bf load_frag_b(const unsigned short* baseT,
                                             int nBase, int stride, int k0) {
  int lane = threadIdx.x & 31;
  int n = lane & 15;
  int kk = k0 + ((lane & 16) >> 1);           // +8 for upper half-wave
  const unsigned short* p = baseT + (size_t)(nBase + n) * stride + kk;
  FragU f;
  f.u[0] = *(const uint4*)(p);                // halves k..k+7
  f.u[1] = *(const uint4*)(p + 16);           // halves k+16..k+23
  return f.v;
}

// ---- prep: transpose + convert weights to bf16 ---------------------------
__global__ void prep_weights(const float* __restrict__ wqkv,
                             const float* __restrict__ wproj,
                             unsigned short* __restrict__ wqkv_t,   // [384][128]
                             unsigned short* __restrict__ wproj_t)  // [128][128]
{
  int idx = blockIdx.x * blockDim.x + threadIdx.x;
  if (idx < 128 * 384) {
    int k = idx / 384, n = idx % 384;          // wqkv is [C=128][3C=384]
    wqkv_t[n * 128 + k] = cvt_bf16(wqkv[idx]);
  }
  if (idx < 128 * 128) {
    int k = idx / 128, n = idx % 128;          // wproj is [128][128]
    wproj_t[n * 128 + k] = cvt_bf16(wproj[idx]);
  }
}

// ---- fused window attention ---------------------------------------------
__global__ __launch_bounds__(256)
void swin_window_attn(const float* __restrict__ x,
                      const float* __restrict__ b_qkv,
                      const float* __restrict__ b_proj,
                      const unsigned short* __restrict__ wqkv_t,
                      const unsigned short* __restrict__ wproj_t,
                      float* __restrict__ out)
{
  extern __shared__ unsigned short lds[];
  unsigned short* X  = lds;            // 64x128 bf16 (padded window)
  unsigned short* Q  = X  + 8192;      // 64x128  [token][chan]
  unsigned short* Kt = Q  + 8192;      // 64x128  [token][chan]
  unsigned short* VT = Kt + 8192;      // 4 heads x [32 x 64]  [d][token]
  unsigned short* P  = VT + 8192;      // 4 heads x [64 x 64]  [query][key]
  unsigned short* O  = P  + 16384;     // 64x128  [token][chan]
  // total 57344 halves = 114688 B

  const int win = blockIdx.x;          // 4096 windows
  const int b   = win >> 8;
  const int wh  = (win >> 4) & 15;
  const int ww  = win & 15;
  const int h0  = wh * 7, w0 = ww * 7;
  const float* xw = x + ((size_t)(b * 112 + h0) * 112 + w0) * 128;

  const int tid  = threadIdx.x;
  const int wave = tid >> 5;
  const int lane = tid & 31;
  const int ln   = lane & 15;
  const int rg   = (lane >> 4) << 3;   // row offset of this lane's D rows

  // ---- phase 0: stage window into LDS as bf16, zero-pad rows 49..63 -----
  {
    int r  = tid >> 2;                 // 0..63
    int c0 = (tid & 3) * 32;
    if (r < 49) {
      const float* src = xw + ((size_t)(r / 7) * 112 + (r % 7)) * 128 + c0;
      #pragma unroll
      for (int c = 0; c < 32; c += 8) {
        float4 f0 = *(const float4*)(src + c);
        float4 f1 = *(const float4*)(src + c + 4);
        uint4 pk;
        pk.x = pack_bf16(f0.x, f0.y);
        pk.y = pack_bf16(f0.z, f0.w);
        pk.z = pack_bf16(f1.x, f1.y);
        pk.w = pack_bf16(f1.z, f1.w);
        *(uint4*)&X[r * 128 + c0 + c] = pk;
      }
    } else {
      #pragma unroll
      for (int c = 0; c < 32; c += 8)
        *(uint4*)&X[r * 128 + c0 + c] = uint4{0u, 0u, 0u, 0u};
    }
  }
  __syncthreads();

  // ---- phase 1: QKV = X @ Wqkv + b ; scatter to Q / Kt / VT -------------
  #pragma unroll
  for (int nti = 0; nti < 3; ++nti) {
    int nt = wave * 3 + nti;           // 24 N-tiles of 16 over 384 cols
    v16bf bf[4];
    #pragma unroll
    for (int ks = 0; ks < 4; ++ks)
      bf[ks] = load_frag_b(wqkv_t, nt * 16, 128, ks * 32);
    int col = nt * 16 + ln;
    float bias = b_qkv[col];
    #pragma unroll
    for (int mt = 0; mt < 4; ++mt) {
      v16bf a[4];
      #pragma unroll
      for (int ks = 0; ks < 4; ++ks)
        a[ks] = load_frag_a(X, mt * 16, 128, ks * 32);
      v8f acc;
      #pragma unroll
      for (int i = 0; i < 8; ++i) acc[i] = bias;
      #pragma unroll
      for (int ks = 0; ks < 4; ++ks)
        acc = __builtin_amdgcn_wmma_f32_16x16x32_bf16(false, a[ks], false, bf[ks],
                                                      (short)0, acc, false, false);
      if (nt < 8) {                    // Q: [token][chan]
        #pragma unroll
        for (int r = 0; r < 8; ++r)
          Q[(mt * 16 + rg + r) * 128 + col] = cvt_bf16(acc[r]);
      } else if (nt < 16) {            // K: [token][chan]
        int c = col - 128;
        #pragma unroll
        for (int r = 0; r < 8; ++r)
          Kt[(mt * 16 + rg + r) * 128 + c] = cvt_bf16(acc[r]);
      } else {                         // V transposed: VT[head][d][token]
        int c = col - 256;
        int head = c >> 5, d = c & 31;
        uint4 pk;                      // 8 consecutive tokens -> one b128 store
        pk.x = pack_bf16(acc[0], acc[1]);
        pk.y = pack_bf16(acc[2], acc[3]);
        pk.z = pack_bf16(acc[4], acc[5]);
        pk.w = pack_bf16(acc[6], acc[7]);
        *(uint4*)&VT[head * 2048 + d * 64 + mt * 16 + rg] = pk;
      }
    }
  }
  __syncthreads();

  // ---- phase 2: S^T = K @ Q^T ; softmax over keys ; P[query][key] -------
  {
    const int head = wave >> 1;        // 2 waves per head
    const int half = wave & 1;
    const float scale = 0.17677669529663687f;   // 1/sqrt(32)
    #pragma unroll
    for (int si = 0; si < 2; ++si) {
      int strip = half * 2 + si;       // query tile 0..3
      v16bf bq = load_frag_b(Q, strip * 16, 128, head * 32);
      v16bf ak[4];
      #pragma unroll
      for (int mt = 0; mt < 4; ++mt)
        ak[mt] = load_frag_a(Kt, mt * 16, 128, head * 32);
      v8f s[4];                        // rows = keys, col = this lane's query
      #pragma unroll
      for (int mt = 0; mt < 4; ++mt) {
        v8f z = {0.f, 0.f, 0.f, 0.f, 0.f, 0.f, 0.f, 0.f};
        s[mt] = __builtin_amdgcn_wmma_f32_16x16x32_bf16(false, ak[mt], false, bq,
                                                        (short)0, z, false, false);
      }
      // softmax over 64 keys of query q (32 keys here, 32 in lane^16)
      float mx = -3.0e38f;
      #pragma unroll
      for (int mt = 0; mt < 4; ++mt)
        #pragma unroll
        for (int r = 0; r < 8; ++r) {
          int key = mt * 16 + rg + r;
          float v = s[mt][r] * scale;
          s[mt][r] = v;
          if (key < 49) mx = fmaxf(mx, v);
        }
      mx = fmaxf(mx, __shfl_xor(mx, 16, 32));
      float sum = 0.f;
      #pragma unroll
      for (int mt = 0; mt < 4; ++mt)
        #pragma unroll
        for (int r = 0; r < 8; ++r) {
          int key = mt * 16 + rg + r;
          float p = (key < 49) ? __expf(s[mt][r] - mx) : 0.f;
          s[mt][r] = p;
          sum += p;
        }
      sum += __shfl_xor(sum, 16, 32);
      float inv = 1.f / sum;
      int q = strip * 16 + ln;
      unsigned base = head * 4096 + q * 64 + rg;   // keys contiguous
      #pragma unroll
      for (int mt = 0; mt < 4; ++mt) {
        uint4 pk;
        pk.x = pack_bf16(s[mt][0] * inv, s[mt][1] * inv);
        pk.y = pack_bf16(s[mt][2] * inv, s[mt][3] * inv);
        pk.z = pack_bf16(s[mt][4] * inv, s[mt][5] * inv);
        pk.w = pack_bf16(s[mt][6] * inv, s[mt][7] * inv);
        *(uint4*)&P[base + mt * 16] = pk;
      }
    }
  }
  __syncthreads();                     // P rows are consumed across waves

  // ---- phase 3: O^T = V^T @ P^T ; store O[token][chan] ------------------
  {
    const int head = wave >> 1;
    const int half = wave & 1;         // which 16 of the 32 head dims
    v16bf av[2];
    #pragma unroll
    for (int ks = 0; ks < 2; ++ks)
      av[ks] = load_frag_a(VT + head * 2048, half * 16, 64, ks * 32);
    #pragma unroll
    for (int nt = 0; nt < 4; ++nt) {   // query tiles
      v16bf bp[2];
      #pragma unroll
      for (int ks = 0; ks < 2; ++ks)
        bp[ks] = load_frag_b(P + head * 4096, nt * 16, 64, ks * 32);
      v8f acc = {0.f, 0.f, 0.f, 0.f, 0.f, 0.f, 0.f, 0.f};
      #pragma unroll
      for (int ks = 0; ks < 2; ++ks)
        acc = __builtin_amdgcn_wmma_f32_16x16x32_bf16(false, av[ks], false, bp[ks],
                                                      (short)0, acc, false, false);
      int q = nt * 16 + ln;            // token
      unsigned base = q * 128 + head * 32 + half * 16 + rg;  // chans contiguous
      uint4 pk;
      pk.x = pack_bf16(acc[0], acc[1]);
      pk.y = pack_bf16(acc[2], acc[3]);
      pk.z = pack_bf16(acc[4], acc[5]);
      pk.w = pack_bf16(acc[6], acc[7]);
      *(uint4*)&O[base] = pk;
    }
  }
  __syncthreads();

  // ---- phase 4: out = O @ Wproj + b ; scatter to global -----------------
  {
    int nt = wave;                      // 8 N-tiles of 16 over 128 cols
    v16bf bw[4];
    #pragma unroll
    for (int ks = 0; ks < 4; ++ks)
      bw[ks] = load_frag_b(wproj_t, nt * 16, 128, ks * 32);
    int col = nt * 16 + ln;
    float bias = b_proj[col];
    #pragma unroll
    for (int mt = 0; mt < 4; ++mt) {
      v16bf a[4];
      #pragma unroll
      for (int ks = 0; ks < 4; ++ks)
        a[ks] = load_frag_a(O, mt * 16, 128, ks * 32);
      v8f acc;
      #pragma unroll
      for (int i = 0; i < 8; ++i) acc[i] = bias;
      #pragma unroll
      for (int ks = 0; ks < 4; ++ks)
        acc = __builtin_amdgcn_wmma_f32_16x16x32_bf16(false, a[ks], false, bw[ks],
                                                      (short)0, acc, false, false);
      #pragma unroll
      for (int r = 0; r < 8; ++r) {
        int row = mt * 16 + rg + r;
        if (row < 49) {
          int i = row / 7, j = row % 7;
          out[((size_t)(b * 112 + h0 + i) * 112 + (w0 + j)) * 128 + col] = acc[r];
        }
      }
    }
  }
}

extern "C" void kernel_launch(void* const* d_in, const int* in_sizes, int n_in,
                              void* d_out, int out_size, void* d_ws, size_t ws_size,
                              hipStream_t stream) {
  (void)in_sizes; (void)n_in; (void)out_size; (void)ws_size;
  const float* x      = (const float*)d_in[0];
  const float* w_qkv  = (const float*)d_in[1];
  const float* b_qkv  = (const float*)d_in[2];
  const float* w_proj = (const float*)d_in[3];
  const float* b_proj = (const float*)d_in[4];
  float* out = (float*)d_out;

  unsigned short* wqkv_t  = (unsigned short*)d_ws;         // 384*128 bf16
  unsigned short* wproj_t = wqkv_t + 384 * 128;            // 128*128 bf16

  prep_weights<<<192, 256, 0, stream>>>(w_qkv, w_proj, wqkv_t, wproj_t);

  const size_t shmem = 57344 * sizeof(unsigned short);     // 114688 B
  swin_window_attn<<<4096, 256, shmem, stream>>>(x, b_qkv, b_proj,
                                                 wqkv_t, wproj_t, out);
}
